// RoutedExpert_43774306681265
// MI455X (gfx1250) — matrix operand order repaired and checked by reading, not compiled, not run on hardware
//
#include <hip/hip_runtime.h>
#include <hip/hip_bf16.h>

typedef __bf16 bf16;
typedef __attribute__((ext_vector_type(16))) __bf16 v16bf;
typedef __attribute__((ext_vector_type(8)))  __bf16 bf16x8;
typedef __attribute__((ext_vector_type(4)))  __bf16 bf16x4;
typedef __attribute__((ext_vector_type(8)))  float  v8f;

#define T_TOK   2048
#define DM      1024
#define HE      512
#define NE      16
#define TK      4
#define MAXROWS 9216   // 8192 assignments + <=63 pad per expert, rounded up

union V16U { v16bf v; bf16x8 h[2]; };

// ---------------------------------------------------------------- init
__global__ void k_init(int* counts, int* arow_tok, float* arow_gate) {
    int i = blockIdx.x * blockDim.x + threadIdx.x;
    if (i < NE) counts[i] = 0;
    if (i < MAXROWS) { arow_tok[i] = 0; arow_gate[i] = 0.0f; }
}

// ---------------------------------------------------------------- router
__launch_bounds__(128)
__global__ void k_router(const float* __restrict__ x, const float* __restrict__ cent,
                         const float* __restrict__ bias,
                         bf16* __restrict__ xb, float* __restrict__ gates4,
                         int* __restrict__ eidx4, int* __restrict__ counts) {
    __shared__ float xr[DM];
    __shared__ float ps[NE][8];
    __shared__ float sc[NE];
    const int t = blockIdx.x, tid = threadIdx.x;

    // load x row, stage to LDS, emit bf16 copy
    float4 v0 = *(const float4*)(x + (size_t)t * DM + tid * 8);
    float4 v1 = *(const float4*)(x + (size_t)t * DM + tid * 8 + 4);
    *(float4*)&xr[tid * 8]     = v0;
    *(float4*)&xr[tid * 8 + 4] = v1;
    bf16x8 hv;
    hv[0]=(bf16)v0.x; hv[1]=(bf16)v0.y; hv[2]=(bf16)v0.z; hv[3]=(bf16)v0.w;
    hv[4]=(bf16)v1.x; hv[5]=(bf16)v1.y; hv[6]=(bf16)v1.z; hv[7]=(bf16)v1.w;
    *(bf16x8*)(xb + (size_t)t * DM + tid * 8) = hv;
    __syncthreads();

    // 16 experts x 8 chunks of 128
    const int e = tid & 15, ch = tid >> 4;
    const float* ce = cent + (size_t)e * DM + ch * 128;
    const float* xp = &xr[ch * 128];
    float s = 0.0f;
    #pragma unroll 4
    for (int j = 0; j < 128; ++j) s += xp[j] * ce[j];
    ps[e][ch] = s;
    __syncthreads();

    if (tid < NE) {
        float v = bias[tid];
        #pragma unroll
        for (int c = 0; c < 8; ++c) v += ps[tid][c];
        sc[tid] = 1.0f / (1.0f + __expf(-v));   // sigmoid
    }
    __syncthreads();

    if (tid == 0) {
        float s2[NE];
        #pragma unroll
        for (int i = 0; i < NE; ++i) s2[i] = sc[i];
        float gv[TK]; int gi[TK]; float gsum = 0.0f;
        for (int k = 0; k < TK; ++k) {                    // top-k, first-occurrence ties
            int best = 0; float bv = s2[0];
            for (int i = 1; i < NE; ++i) if (s2[i] > bv) { bv = s2[i]; best = i; }
            gv[k] = bv; gi[k] = best; s2[best] = -1e30f; gsum += bv;
        }
        gsum = fmaxf(gsum, 1e-9f);
        for (int k = 0; k < TK; ++k) {
            gates4[t * TK + k] = gv[k] / gsum;
            eidx4[t * TK + k]  = gi[k];
            atomicAdd(&counts[gi[k]], 1);
        }
    }
}

// ---------------------------------------------------------------- scan (E=16, serial)
__global__ void k_scan(const int* counts, int* poff, int* pcount, int* cursor) {
    if (threadIdx.x == 0 && blockIdx.x == 0) {
        int off = 0;
        for (int e = 0; e < NE; ++e) {
            int c  = counts[e];
            int pc = (c + 63) & ~63;       // pad to 64-row tiles
            poff[e] = off; pcount[e] = pc; cursor[e] = off;
            off += pc;
        }
    }
}

// ---------------------------------------------------------------- fill assignment rows
__global__ void k_fill(const float* __restrict__ gates4, const int* __restrict__ eidx4,
                       int* cursor, int* arow_tok, float* arow_gate, int* inv) {
    int t = blockIdx.x * blockDim.x + threadIdx.x;
    if (t >= T_TOK) return;
    for (int k = 0; k < TK; ++k) {
        int e   = eidx4[t * TK + k];
        int pos = atomicAdd(&cursor[e], 1);
        arow_tok[pos]  = t;
        arow_gate[pos] = gates4[t * TK + k];
        inv[t * TK + k] = pos;
    }
}

// ---------------------------------------------------------------- fp32 -> bf16, transposed
// src: [NE][R][C] fp32 (row-major). dst: [NE][C][R] bf16 (N-major, K-contiguous).
// 32x32 LDS tile: coalesced f32 reads, coalesced b64 bf16 writes.
__launch_bounds__(256)
__global__ void k_cvt_t(const float* __restrict__ src, bf16* __restrict__ dst,
                        int R, int C) {
    __shared__ bf16 tile[32][36];
    const int e  = blockIdx.z;
    const int r0 = blockIdx.y * 32, c0 = blockIdx.x * 32;
    const int tid = threadIdx.x;
    const float* s = src + (size_t)e * R * C;
    bf16*        d = dst + (size_t)e * R * C;

    const int lr = tid >> 3;          // 0..31
    const int lc = (tid & 7) * 4;     // 0,4,..,28
    float4 v = *(const float4*)(s + (size_t)(r0 + lr) * C + c0 + lc);
    tile[lr][lc + 0] = (bf16)v.x;
    tile[lr][lc + 1] = (bf16)v.y;
    tile[lr][lc + 2] = (bf16)v.z;
    tile[lr][lc + 3] = (bf16)v.w;
    __syncthreads();

    const int cr = tid >> 3;          // dst row (= src col)
    const int rc = (tid & 7) * 4;     // dst col group (= src row)
    bf16x4 o;
    o[0] = tile[rc + 0][cr];
    o[1] = tile[rc + 1][cr];
    o[2] = tile[rc + 2][cr];
    o[3] = tile[rc + 3][cr];
    *(bf16x4*)(d + (size_t)(c0 + cr) * R + r0 + rc) = o;
}

// ---------------------------------------------------------------- stage 1: gate/up + SwiGLU
// Block: 64 assignment rows x 128 H-cols, K = DM, dual GEMM sharing A.
// Weights pre-transposed (N-major) -> straight b128 LDS staging, no in-kernel transpose.
__launch_bounds__(256)
__global__ void k_gemm1(const bf16* __restrict__ xb, const bf16* __restrict__ wgb,
                        const bf16* __restrict__ wub,
                        const int* __restrict__ arow_tok, const float* __restrict__ arow_gate,
                        const int* __restrict__ poff, const int* __restrict__ pcount,
                        bf16* __restrict__ hb) {
    const int e     = blockIdx.z;
    const int mtile = blockIdx.y;
    const int nbase = blockIdx.x * 128;
    const int pc    = pcount[e];
    if (mtile * 64 >= pc) return;
    const int rowbase = poff[e] + mtile * 64;

    __shared__ __align__(16) bf16 As[2][64][72];
    __shared__ __align__(16) bf16 Bg[2][128][72];   // [n][k]
    __shared__ __align__(16) bf16 Bu[2][128][72];
    __shared__ __align__(16) bf16 Ho[64][136];      // output staging
    __shared__ float gts[64];
    __shared__ int   tks[64];

    const int tid  = threadIdx.x;
    const int lane = tid & 31;
    const int wav  = tid >> 5;
    const int m0w  = (wav >> 2) * 32;    // 2 wave-rows
    const int n0w  = (wav & 3)  * 32;    // 4 wave-cols

    if (tid < 64) {
        tks[tid] = arow_tok[rowbase + tid];
        gts[tid] = arow_gate[rowbase + tid];
    }
    __syncthreads();

    v8f accg[2][2], accu[2][2];
    const v8f vzero = {0,0,0,0,0,0,0,0};
    #pragma unroll
    for (int i = 0; i < 2; ++i)
        #pragma unroll
        for (int j = 0; j < 2; ++j) { accg[i][j] = vzero; accu[i][j] = vzero; }

    const bf16* wg_e = wgb + (size_t)e * DM * HE;   // [HE][DM]
    const bf16* wu_e = wub + (size_t)e * DM * HE;   // [HE][DM]

    const int arow = tid >> 2;            // 0..63
    const int aseg = (tid & 3) * 16;      // 16 bf16 per thread
    const int bnr  = tid >> 1;            // 0..127  B row (n)
    const int bks  = (tid & 1) * 32;      // k offset: 32 bf16 per thread

    bf16x8 rA0, rA1, rBg[4], rBu[4];

    auto g2r = [&](int k0) {
        const bf16* srcA = xb + (size_t)tks[arow] * DM + k0 + aseg;
        rA0 = *(const bf16x8*)srcA;
        rA1 = *(const bf16x8*)(srcA + 8);
        const bf16* sg = wg_e + (size_t)(nbase + bnr) * DM + k0 + bks;
        const bf16* su = wu_e + (size_t)(nbase + bnr) * DM + k0 + bks;
        #pragma unroll
        for (int p = 0; p < 4; ++p) {
            rBg[p] = *(const bf16x8*)(sg + p * 8);
            rBu[p] = *(const bf16x8*)(su + p * 8);
        }
    };
    auto r2l = [&](int b) {
        *(bf16x8*)&As[b][arow][aseg]     = rA0;
        *(bf16x8*)&As[b][arow][aseg + 8] = rA1;
        #pragma unroll
        for (int p = 0; p < 4; ++p) {
            *(bf16x8*)&Bg[b][bnr][bks + p * 8] = rBg[p];
            *(bf16x8*)&Bu[b][bnr][bks + p * 8] = rBu[p];
        }
    };
    auto compute = [&](int b) {
        #pragma unroll
        for (int ks = 0; ks < 64; ks += 32) {
            const int khA = ks + (lane >> 4) * 16;    // A: K 0..15 / 16..31
            const int khB = ks + (lane >> 4) * 16;    // B: K 0..15 / 16..31
            v16bf a[2];
            #pragma unroll
            for (int mi = 0; mi < 2; ++mi) {
                const bf16* ap = &As[b][m0w + mi * 16 + (lane & 15)][khA];
                V16U u; u.h[0] = *(const bf16x8*)ap; u.h[1] = *(const bf16x8*)(ap + 8);
                a[mi] = u.v;
            }
            #pragma unroll
            for (int ni = 0; ni < 2; ++ni) {
                const bf16* gp = &Bg[b][n0w + ni * 16 + (lane & 15)][khB];
                const bf16* up = &Bu[b][n0w + ni * 16 + (lane & 15)][khB];
                V16U ug; ug.h[0] = *(const bf16x8*)gp; ug.h[1] = *(const bf16x8*)(gp + 8);
                V16U uu; uu.h[0] = *(const bf16x8*)up; uu.h[1] = *(const bf16x8*)(up + 8);
                #pragma unroll
                for (int mi = 0; mi < 2; ++mi) {
                    accg[mi][ni] = __builtin_amdgcn_wmma_f32_16x16x32_bf16(
                        false, a[mi], false, ug.v, (short)0, accg[mi][ni], false, false);
                    accu[mi][ni] = __builtin_amdgcn_wmma_f32_16x16x32_bf16(
                        false, a[mi], false, uu.v, (short)0, accu[mi][ni], false, false);
                }
            }
        }
    };

    g2r(0); r2l(0);
    __syncthreads();
    const int NITER = DM / 64;
    #pragma unroll 2
    for (int i = 0; i < NITER; ++i) {
        const int cur = i & 1;
        if (i + 1 < NITER) g2r((i + 1) * 64);   // next tile: global loads in flight
        compute(cur);                            // WMMAs cover the load latency
        if (i + 1 < NITER) r2l(cur ^ 1);        // park next tile in other buffer
        __syncthreads();
    }

    // epilogue: h = silu(g)*u*gate -> LDS stage -> coalesced b128 stores
    const int ln = lane & 15, lh = lane >> 4;
    #pragma unroll
    for (int mi = 0; mi < 2; ++mi)
        #pragma unroll
        for (int ni = 0; ni < 2; ++ni) {
            const int col = n0w + ni * 16 + ln;
            #pragma unroll
            for (int r = 0; r < 8; ++r) {
                int m = m0w + mi * 16 + r + lh * 8;
                float g = accg[mi][ni][r];
                float u = accu[mi][ni][r];
                Ho[m][col] = (bf16)(g / (1.0f + __expf(-g)) * u * gts[m]);
            }
        }
    __syncthreads();
    {
        const int orow = tid >> 2, oseg = (tid & 3) * 32;
        bf16* dst = hb + (size_t)(rowbase + orow) * HE + nbase + oseg;
        #pragma unroll
        for (int j = 0; j < 4; ++j)
            *(bf16x8*)(dst + j * 8) = *(const bf16x8*)&Ho[orow][oseg + j * 8];
    }
}

// ---------------------------------------------------------------- stage 2: down proj
__launch_bounds__(256)
__global__ void k_gemm2(const bf16* __restrict__ hb, const bf16* __restrict__ wdb,
                        const int* __restrict__ poff, const int* __restrict__ pcount,
                        bf16* __restrict__ yb) {
    const int e     = blockIdx.z;
    const int mtile = blockIdx.y;
    const int nbase = blockIdx.x * 128;
    const int pc    = pcount[e];
    if (mtile * 64 >= pc) return;
    const int rowbase = poff[e] + mtile * 64;

    __shared__ __align__(16) bf16 As[2][64][72];
    __shared__ __align__(16) bf16 Bs[2][128][72];
    __shared__ __align__(16) bf16 Yo[64][136];

    const int tid  = threadIdx.x;
    const int lane = tid & 31;
    const int wav  = tid >> 5;
    const int m0w  = (wav >> 2) * 32;
    const int n0w  = (wav & 3)  * 32;

    v8f acc[2][2];
    const v8f vzero = {0,0,0,0,0,0,0,0};
    #pragma unroll
    for (int i = 0; i < 2; ++i)
        #pragma unroll
        for (int j = 0; j < 2; ++j) acc[i][j] = vzero;

    const bf16* wd_e = wdb + (size_t)e * HE * DM;   // [DM][HE]
    const int arow = tid >> 2;
    const int aseg = (tid & 3) * 16;
    const int bnr  = tid >> 1;            // 0..127
    const int bks  = (tid & 1) * 32;

    bf16x8 rA0, rA1, rB[4];

    auto g2r = [&](int k0) {
        const bf16* srcA = hb + (size_t)(rowbase + arow) * HE + k0 + aseg;
        rA0 = *(const bf16x8*)srcA;
        rA1 = *(const bf16x8*)(srcA + 8);
        const bf16* sb = wd_e + (size_t)(nbase + bnr) * HE + k0 + bks;
        #pragma unroll
        for (int p = 0; p < 4; ++p) rB[p] = *(const bf16x8*)(sb + p * 8);
    };
    auto r2l = [&](int b) {
        *(bf16x8*)&As[b][arow][aseg]     = rA0;
        *(bf16x8*)&As[b][arow][aseg + 8] = rA1;
        #pragma unroll
        for (int p = 0; p < 4; ++p) *(bf16x8*)&Bs[b][bnr][bks + p * 8] = rB[p];
    };
    auto compute = [&](int b) {
        #pragma unroll
        for (int ks = 0; ks < 64; ks += 32) {
            const int khA = ks + (lane >> 4) * 16;
            const int khB = ks + (lane >> 4) * 16;
            v16bf a[2];
            #pragma unroll
            for (int mi = 0; mi < 2; ++mi) {
                const bf16* ap = &As[b][m0w + mi * 16 + (lane & 15)][khA];
                V16U u; u.h[0] = *(const bf16x8*)ap; u.h[1] = *(const bf16x8*)(ap + 8);
                a[mi] = u.v;
            }
            #pragma unroll
            for (int ni = 0; ni < 2; ++ni) {
                const bf16* bp = &Bs[b][n0w + ni * 16 + (lane & 15)][khB];
                V16U ub; ub.h[0] = *(const bf16x8*)bp; ub.h[1] = *(const bf16x8*)(bp + 8);
                #pragma unroll
                for (int mi = 0; mi < 2; ++mi)
                    acc[mi][ni] = __builtin_amdgcn_wmma_f32_16x16x32_bf16(
                        false, a[mi], false, ub.v, (short)0, acc[mi][ni], false, false);
            }
        }
    };

    g2r(0); r2l(0);
    __syncthreads();
    const int NITER = HE / 64;
    #pragma unroll 2
    for (int i = 0; i < NITER; ++i) {
        const int cur = i & 1;
        if (i + 1 < NITER) g2r((i + 1) * 64);
        compute(cur);
        if (i + 1 < NITER) r2l(cur ^ 1);
        __syncthreads();
    }

    const int ln = lane & 15, lh = lane >> 4;
    #pragma unroll
    for (int mi = 0; mi < 2; ++mi)
        #pragma unroll
        for (int ni = 0; ni < 2; ++ni) {
            const int col = n0w + ni * 16 + ln;
            #pragma unroll
            for (int r = 0; r < 8; ++r) {
                int m = m0w + mi * 16 + r + lh * 8;
                Yo[m][col] = (bf16)acc[mi][ni][r];
            }
        }
    __syncthreads();
    {
        const int orow = tid >> 2, oseg = (tid & 3) * 32;
        bf16* dst = yb + (size_t)(rowbase + orow) * DM + nbase + oseg;
        #pragma unroll
        for (int j = 0; j < 4; ++j)
            *(bf16x8*)(dst + j * 8) = *(const bf16x8*)&Yo[orow][oseg + j * 8];
    }
}

// ---------------------------------------------------------------- combine
__global__ void k_combine(const bf16* __restrict__ yb, const int* __restrict__ inv,
                          const float* __restrict__ scale, float* __restrict__ out) {
    const int t = blockIdx.x;
    const int d = threadIdx.x * 4;
    const int4 rr = *(const int4*)(inv + t * TK);
    const int rows[4] = { rr.x, rr.y, rr.z, rr.w };
    float s0 = 0, s1 = 0, s2 = 0, s3 = 0;
    #pragma unroll
    for (int k = 0; k < TK; ++k) {
        bf16x4 v = *(const bf16x4*)(yb + (size_t)rows[k] * DM + d);
        s0 += (float)v[0]; s1 += (float)v[1]; s2 += (float)v[2]; s3 += (float)v[3];
    }
    const float sc = scale[0];
    float4 o; o.x = s0 * sc; o.y = s1 * sc; o.z = s2 * sc; o.w = s3 * sc;
    *(float4*)(out + (size_t)t * DM + d) = o;
}

// ---------------------------------------------------------------- launch
extern "C" void kernel_launch(void* const* d_in, const int* in_sizes, int n_in,
                              void* d_out, int out_size, void* d_ws, size_t ws_size,
                              hipStream_t stream) {
    const float* x      = (const float*)d_in[0];
    const float* cent   = (const float*)d_in[1];
    const float* bias   = (const float*)d_in[2];
    const float* w_gate = (const float*)d_in[3];
    const float* w_up   = (const float*)d_in[4];
    const float* w_down = (const float*)d_in[5];
    const float* scale  = (const float*)d_in[6];
    float* out = (float*)d_out;

    char* p = (char*)d_ws;
    auto alloc = [&](size_t bytes) -> char* {
        char* r = p; p += (bytes + 255) & ~(size_t)255; return r;
    };
    bf16*  xb        = (bf16*)alloc((size_t)T_TOK * DM * 2);
    bf16*  wgb       = (bf16*)alloc((size_t)NE * DM * HE * 2);   // [e][HE][DM]
    bf16*  wub       = (bf16*)alloc((size_t)NE * DM * HE * 2);   // [e][HE][DM]
    bf16*  wdb       = (bf16*)alloc((size_t)NE * HE * DM * 2);   // [e][DM][HE]
    bf16*  hb        = (bf16*)alloc((size_t)MAXROWS * HE * 2);
    bf16*  yb        = (bf16*)alloc((size_t)MAXROWS * DM * 2);
    float* gates4    = (float*)alloc((size_t)T_TOK * TK * 4);
    int*   eidx4     = (int*)alloc((size_t)T_TOK * TK * 4);
    int*   inv       = (int*)alloc((size_t)T_TOK * TK * 4);
    int*   arow_tok  = (int*)alloc((size_t)MAXROWS * 4);
    float* arow_gate = (float*)alloc((size_t)MAXROWS * 4);
    int*   counts    = (int*)alloc(64);
    int*   poff      = (int*)alloc(64);
    int*   pcount    = (int*)alloc(64);
    int*   cursor    = (int*)alloc(64);

    k_init  <<<(MAXROWS + 255) / 256, 256, 0, stream>>>(counts, arow_tok, arow_gate);
    k_router<<<T_TOK, 128, 0, stream>>>(x, cent, bias, xb, gates4, eidx4, counts);
    k_scan  <<<1, 32, 0, stream>>>(counts, poff, pcount, cursor);
    k_fill  <<<(T_TOK + 255) / 256, 256, 0, stream>>>(gates4, eidx4, cursor,
                                                      arow_tok, arow_gate, inv);
    // convert + transpose weights once: bf16, N-major (K contiguous per row)
    k_cvt_t<<<dim3(HE / 32, DM / 32, NE), 256, 0, stream>>>(w_gate, wgb, DM, HE);
    k_cvt_t<<<dim3(HE / 32, DM / 32, NE), 256, 0, stream>>>(w_up,   wub, DM, HE);
    k_cvt_t<<<dim3(DM / 32, HE / 32, NE), 256, 0, stream>>>(w_down, wdb, HE, DM);

    k_gemm1<<<dim3(HE / 128, 32, NE), 256, 0, stream>>>(xb, wgb, wub, arow_tok, arow_gate,
                                                        poff, pcount, hb);
    k_gemm2<<<dim3(DM / 128, 32, NE), 256, 0, stream>>>(hb, wdb, poff, pcount, yb);
    k_combine<<<T_TOK, 256, 0, stream>>>(yb, inv, scale, out);
}
